// Nerf_86981677679028
// MI455X (gfx1250) — matrix operand order repaired and checked by reading, not compiled, not run on hardware
//
#include <hip/hip_runtime.h>
#include <hip/hip_fp16.h>

// ---------------------------------------------------------------------------
// Fused NeRF MLP for MI455X (gfx1250), wave32 + WMMA f32_16x16x32_f16.
// Workgroup = 256 threads (8 waves), processes 128 points.
// Each wave owns a 16-row activation tile in LDS; layer weights are staged
// into a shared LDS buffer (f16, padded) via the Tensor Data Mover (TDM)
// and consumed as WMMA B-fragments.
// ---------------------------------------------------------------------------

typedef __attribute__((ext_vector_type(16))) _Float16 v16h;
typedef __attribute__((ext_vector_type(8)))  _Float16 h8;
typedef __attribute__((ext_vector_type(8)))  float    v8f;
typedef __attribute__((ext_vector_type(4)))  unsigned int u32x4;
typedef __attribute__((ext_vector_type(8)))  int          i32x8;
typedef __attribute__((ext_vector_type(4)))  int          i32x4;

#define NPTS        131072
#define ACT_STRIDE  336          // halves per activation row (>=320, 16B aligned)
#define SW_HALVES   81920        // max layer weight tile: 256 x 320 f16
#define ACT_HALVES  (8 * 16 * ACT_STRIDE)
#define BIAS_BYTE_OFF 1200128    // 600064 halves of packed f16 weights

#if defined(__has_builtin)
#if __has_builtin(__builtin_amdgcn_tensor_load_to_lds) && \
    __has_builtin(__builtin_amdgcn_s_wait_tensorcnt)
#define USE_TDM 1
#endif
#endif

// ---------------- weight pre-pass: f32 -> padded f16 + padded f32 bias -----
__global__ void cvt_layer(const float* __restrict__ W, const float* __restrict__ b,
                          _Float16* __restrict__ wOut, float* __restrict__ bOut,
                          int K, int Kpad, int N, int Npad)
{
    int idx = blockIdx.x * 256 + threadIdx.x;
    int total = Npad * Kpad;
    if (idx < total) {
        int n = idx / Kpad;
        int k = idx - n * Kpad;
        float v = (n < N && k < K) ? W[n * K + k] : 0.0f;
        wOut[idx] = (_Float16)v;
    }
    if (idx < Npad) bOut[idx] = (idx < N) ? b[idx] : 0.0f;
}

// ---------------- fragment loaders -----------------------------------------
// A (16x32 f16): lanes 0-15 row M=lane: halves K0..7 then K16..23
//                lanes 16-31 same rows:  halves K8..15 then K24..31
__device__ __forceinline__ v16h load_fragA(const _Float16* p)
{
    h8 lo = *(const h8*)(p);
    h8 hh = *(const h8*)(p + 16);
    return __builtin_shufflevector(lo, hh, 0,1,2,3,4,5,6,7,8,9,10,11,12,13,14,15);
}
// B (32x16 f16): lane = N (mod 16); lanes 0-15 hold K0..7,K16..23, lanes 16-31 K8..15,K24..31
__device__ __forceinline__ v16h load_fragB(const _Float16* p)
{
    h8 lo = *(const h8*)(p);
    h8 hh = *(const h8*)(p + 16);
    return __builtin_shufflevector(lo, hh, 0,1,2,3,4,5,6,7,8,9,10,11,12,13,14,15);
}

// ---------------- weight staging: global f16 -> LDS -------------------------
// TDM path: one tensor_load_to_lds describes the whole Npad x Kpad f16 tile
// as a 2D tensor (dim0 = Kpad elems, stride Kpad, dim1 = Npad rows).
// Issued by wave 0 only; completion via s_wait_tensorcnt, published by barrier.
__device__ __forceinline__ void stage_weights(_Float16* sW, const _Float16* g,
                                              int Kpad, int Npad,
                                              int tid, int wave)
{
    __syncthreads();                       // previous layer done reading sW
#if defined(USE_TDM)
    if (wave == 0) {
        unsigned long long ga = (unsigned long long)(size_t)g;
        unsigned int ldsOff   = (unsigned int)(size_t)sW;  // LDS aperture: low 32 bits
        u32x4 g0 = { 1u,                                   // count=1, user descriptor
                     ldsOff,                               // lds_addr
                     (unsigned int)ga,                     // global_addr[31:0]
                     (unsigned int)((ga >> 32) & 0x1FFFFFFull) | (2u << 30) }; // [56:32] | type=2
        i32x8 g1 = { (int)0x00010000,                      // wg_mask=0, data_size=2B
                     (int)((unsigned)(Kpad & 0xFFFF) << 16),   // tensor_dim0[15:0]
                     (int)((unsigned)(Npad & 0xFFFF) << 16),   // dim0 hi | tensor_dim1[15:0]
                     (int)((unsigned)(Kpad & 0xFFFF) << 16),   // dim1 hi | tile_dim0
                     (int)(Npad & 0xFFFF),                     // tile_dim1 (tile_dim2=0)
                     Kpad,                                     // tensor_dim0_stride[31:0]
                     0, 0 };                                   // stride hi, dim1_stride
        i32x4 gz = { 0, 0, 0, 0 };
#if __clang_major__ >= 23
        i32x8 gz8 = { 0, 0, 0, 0, 0, 0, 0, 0 };
        __builtin_amdgcn_tensor_load_to_lds(g0, g1, gz, gz, gz8, 0);
#else
        __builtin_amdgcn_tensor_load_to_lds(g0, g1, gz, gz, 0);
#endif
        __builtin_amdgcn_s_wait_tensorcnt(0);
    }
#else
    {
        const float4* src = (const float4*)g;  // 16B chunks, offsets 16B aligned
        float4* dst = (float4*)sW;
        const int n16 = (Kpad * Npad) >> 3;
        for (int i = tid; i < n16; i += 256) dst[i] = src[i];
    }
#endif
    __syncthreads();                       // weights visible to all waves
}

// ---------------- one dense layer: act[16xKpad] @ W^T -> act[16xNT*16] ------
template<int KCH, int NT, bool RELU>
__device__ void dense_layer(const _Float16* __restrict__ sW,
                            const float* __restrict__ bias,
                            _Float16* __restrict__ actW, int lane)
{
    const int r     = lane & 15;
    const int hi    = lane >> 4;
    const int mbase = hi * 8;

    v16h a[KCH];
    {
        const _Float16* ar = actW + r * ACT_STRIDE + hi * 8;
        #pragma unroll
        for (int kc = 0; kc < KCH; ++kc) a[kc] = load_fragA(ar + kc * 32);
    }
    for (int t = 0; t < NT; ++t) {
        const int n = t * 16 + r;
        const float bv = bias[n];
        v8f c = {bv, bv, bv, bv, bv, bv, bv, bv};
        const _Float16* wr = sW + n * (KCH * 32) + hi * 8;
        #pragma unroll
        for (int kc = 0; kc < KCH; ++kc) {
            v16h b = load_fragB(wr + kc * 32);
            c = __builtin_amdgcn_wmma_f32_16x16x32_f16(false, a[kc], false, b,
                                                       (short)0, c, false, false);
        }
        #pragma unroll
        for (int v = 0; v < 8; ++v) {
            float x = c[v];
            if (RELU) x = fmaxf(x, 0.0f);
            actW[(mbase + v) * ACT_STRIDE + n] = (_Float16)x;
        }
    }
}

// ---------------- fused forward kernel --------------------------------------
__global__ void __launch_bounds__(256)
nerf_fused(const float* __restrict__ xyz, const float* __restrict__ dirs,
           const _Float16* __restrict__ wAll, const float* __restrict__ bAll,
           float* __restrict__ outC, float* __restrict__ outSigma)
{
    extern __shared__ _Float16 smem[];
    _Float16* sW   = smem;                 // SW_HALVES
    _Float16* sAct = smem + SW_HALVES;     // 8 waves x 16 x ACT_STRIDE

    const int tid  = threadIdx.x;
    const int wave = tid >> 5;
    const int lane = tid & 31;
    const int r    = lane & 15;
    const int hi   = lane >> 4;
    const int mbase = hi * 8;
    _Float16* actW = sAct + wave * 16 * ACT_STRIDE;
    const int rowbase = blockIdx.x * 128 + wave * 16;

    // ---- positional encoding of xyz: cols [0..63] and duplicate [256..319]
    {
        float xv[3];
        #pragma unroll
        for (int d = 0; d < 3; ++d)
            xv[d] = fminf(fmaxf(xyz[(rowbase + r) * 3 + d], -5.0f), 5.0f);
        _Float16* rowp = actW + r * ACT_STRIDE;
        for (int c = hi * 32; c < hi * 32 + 32; ++c) {
            float val = 0.0f;
            if (c < 3) val = xv[c];
            else if (c < 51) {
                int j = c - 3;
                int l = j / 6, rem = j % 6;
                float ang = xv[rem % 3] * (float)(1 << l);
                float sc = (rem / 3) ? __cosf(ang) : __sinf(ang);
                val = fminf(fmaxf(sc, -1.0f), 1.0f);
            }
            _Float16 hv = (_Float16)val;
            rowp[c] = hv;
            rowp[256 + c] = hv;   // skip-connection copy for block2 layer 0
        }
    }

    // ---- block1: 51->256, then 4 x 256->256 (ReLU) ----
    stage_weights(sW, wAll + 0,      64, 256, tid, wave);
    dense_layer<2, 16, true>(sW, bAll + 0,    actW, lane);
    stage_weights(sW, wAll + 16384,  256, 256, tid, wave);
    dense_layer<8, 16, true>(sW, bAll + 256,  actW, lane);
    stage_weights(sW, wAll + 81920,  256, 256, tid, wave);
    dense_layer<8, 16, true>(sW, bAll + 512,  actW, lane);
    stage_weights(sW, wAll + 147456, 256, 256, tid, wave);
    dense_layer<8, 16, true>(sW, bAll + 768,  actW, lane);
    stage_weights(sW, wAll + 212992, 256, 256, tid, wave);
    dense_layer<8, 16, true>(sW, bAll + 1024, actW, lane);

    // ---- block2: [h, x_emb] (320) -> 256, 2 x 256->256 (ReLU) ----
    stage_weights(sW, wAll + 278528, 320, 256, tid, wave);
    dense_layer<10, 16, true>(sW, bAll + 1280, actW, lane);
    stage_weights(sW, wAll + 360448, 256, 256, tid, wave);
    dense_layer<8, 16, true>(sW, bAll + 1536, actW, lane);
    stage_weights(sW, wAll + 425984, 256, 256, tid, wave);
    dense_layer<8, 16, true>(sW, bAll + 1792, actW, lane);

    // ---- block2 final: 256 -> 257 (linear). Sigma tile first, then feat.
    stage_weights(sW, wAll + 491520, 256, 272, tid, wave);
    {   // output column 256 (sigma), n = 256 + r (only n==256 is real)
        v16h a[8];
        const _Float16* ar = actW + r * ACT_STRIDE + hi * 8;
        #pragma unroll
        for (int kc = 0; kc < 8; ++kc) a[kc] = load_fragA(ar + kc * 32);
        const int n = 256 + r;
        const float bv = bAll[2048 + n];
        v8f c = {bv, bv, bv, bv, bv, bv, bv, bv};
        const _Float16* wr = sW + n * 256 + hi * 8;
        #pragma unroll
        for (int kc = 0; kc < 8; ++kc) {
            v16h b = load_fragB(wr + kc * 32);
            c = __builtin_amdgcn_wmma_f32_16x16x32_f16(false, a[kc], false, b,
                                                       (short)0, c, false, false);
        }
        if (r == 0) {
            #pragma unroll
            for (int v = 0; v < 8; ++v)
                outSigma[rowbase + mbase + v] = fminf(fmaxf(c[v], 0.0f), 10.0f);
        }
    }
    dense_layer<8, 16, false>(sW, bAll + 2048, actW, lane);  // feat, no ReLU

    // ---- directional encoding into cols [256..287] (feat stays in 0..255)
    {
        float dv[3];
        #pragma unroll
        for (int d = 0; d < 3; ++d) dv[d] = dirs[(rowbase + r) * 3 + d];
        float nrm = sqrtf(dv[0]*dv[0] + dv[1]*dv[1] + dv[2]*dv[2]);
        float inv = 1.0f / fmaxf(nrm, 1e-12f);
        #pragma unroll
        for (int d = 0; d < 3; ++d) dv[d] *= inv;
        _Float16* rowp = actW + r * ACT_STRIDE + 256;
        for (int c = hi * 16; c < hi * 16 + 16; ++c) {
            float val = 0.0f;
            if (c < 3) val = dv[c];
            else if (c < 21) {
                int j = c - 3;
                int l = j / 6, rem = j % 6;
                float ang = dv[rem % 3] * (float)(1 << l);
                float sc = (rem / 3) ? __cosf(ang) : __sinf(ang);
                val = fminf(fmaxf(sc, -1.0f), 1.0f);
            }
            rowp[c] = (_Float16)val;
        }
    }

    // ---- rgb: [feat, d_emb] (288) -> 128 (ReLU) ----
    stage_weights(sW, wAll + 561152, 288, 128, tid, wave);
    dense_layer<9, 8, true>(sW, bAll + 2320, actW, lane);

    // ---- rgb: 128 -> 3 (sigmoid), single output tile ----
    stage_weights(sW, wAll + 598016, 128, 16, tid, wave);
    {
        v16h a[4];
        const _Float16* ar = actW + r * ACT_STRIDE + hi * 8;
        #pragma unroll
        for (int kc = 0; kc < 4; ++kc) a[kc] = load_fragA(ar + kc * 32);
        const float bv = bAll[2448 + r];
        v8f c = {bv, bv, bv, bv, bv, bv, bv, bv};
        const _Float16* wr = sW + r * 128 + hi * 8;
        #pragma unroll
        for (int kc = 0; kc < 4; ++kc) {
            v16h b = load_fragB(wr + kc * 32);
            c = __builtin_amdgcn_wmma_f32_16x16x32_f16(false, a[kc], false, b,
                                                       (short)0, c, false, false);
        }
        if (r < 3) {
            #pragma unroll
            for (int v = 0; v < 8; ++v) {
                float s = 1.0f / (1.0f + __expf(-c[v]));
                s = fminf(fmaxf(s, 0.0f), 1.0f);
                outC[(rowbase + mbase + v) * 3 + r] = s;
            }
        }
    }
}

// ---------------------------------------------------------------------------
extern "C" void kernel_launch(void* const* d_in, const int* in_sizes, int n_in,
                              void* d_out, int out_size, void* d_ws, size_t ws_size,
                              hipStream_t stream)
{
    (void)in_sizes; (void)n_in; (void)out_size; (void)ws_size;
    const float* xyz  = (const float*)d_in[0];
    const float* dirs = (const float*)d_in[1];
    _Float16* wAll = (_Float16*)d_ws;
    float*    bAll = (float*)((char*)d_ws + BIAS_BYTE_OFF);

    // layer geometry: K actual, K padded (mult of 32), N actual, N padded (mult of 16)
    static const int LK [11] = {51, 256, 256, 256, 256, 307, 256, 256, 256, 277, 128};
    static const int LKP[11] = {64, 256, 256, 256, 256, 320, 256, 256, 256, 288, 128};
    static const int LN [11] = {256,256, 256, 256, 256, 256, 256, 256, 257, 128, 3  };
    static const int LNP[11] = {256,256, 256, 256, 256, 256, 256, 256, 272, 128, 16 };

    int woff = 0, boff = 0;
    for (int i = 0; i < 11; ++i) {
        const float* W = (const float*)d_in[2 + 2 * i];
        const float* b = (const float*)d_in[3 + 2 * i];
        int total = LNP[i] * LKP[i];
        int grid  = (total + 255) / 256;
        cvt_layer<<<grid, 256, 0, stream>>>(W, b, wAll + woff, bAll + boff,
                                            LK[i], LKP[i], LN[i], LNP[i]);
        woff += total;
        boff += LNP[i];
    }

    float* outC     = (float*)d_out;
    float* outSigma = outC + (size_t)NPTS * 3;
    size_t smemBytes = (size_t)(SW_HALVES + ACT_HALVES) * sizeof(_Float16);
    nerf_fused<<<NPTS / 128, 256, smemBytes, stream>>>(xyz, dirs, wAll, bAll,
                                                       outC, outSigma);
}
